// RadixAttention_56332791054779
// MI455X (gfx1250) — matrix operand order repaired and checked by reading, not compiled, not run on hardware
//
#include <hip/hip_runtime.h>

typedef __attribute__((ext_vector_type(16))) __bf16 bf16x16;
typedef __attribute__((ext_vector_type(8)))  float  f32x8;

#define NB    4
#define NE    512
#define NP    2048
#define NHQ   32
#define NHKV  8
#define ND    128
#define NG    4
#define QSCALE 0.08838834764831845f

#define KT    64     // keys per LDS tile
#define QBLK  32     // queries per workgroup (x4 q-heads = 8 wave-tiles)
#define KROW  136    // sK row stride (bf16 elems): 128 + 8 pad -> 272B, conflict-free b128
#define VROW  72     // sVT row stride (bf16 elems): 64 + 8 pad  -> 144B
#define PROW  40     // sP row stride (bf16 elems): 32 + 8 pad   -> 80B

// butterfly shuffle within 16-lane halves (group-of-32 swizzle: and=0x1F, xor=m)
#define SWZ(x,m) __int_as_float(__builtin_amdgcn_ds_swizzle(__float_as_int(x), (((m)<<10)|0x1F)))

__device__ __forceinline__ unsigned short f2bf(float f){
  unsigned u = __float_as_uint(f);
  u += 0x7FFFu + ((u >> 16) & 1u);      // round-to-nearest-even
  return (unsigned short)(u >> 16);
}

#if defined(__has_builtin) && __has_builtin(__builtin_amdgcn_cvt_pk_bf16_f32)
__device__ __forceinline__ unsigned pk2(float a, float b){
  auto r = __builtin_amdgcn_cvt_pk_bf16_f32(a, b);   // v_cvt_pk_bf16_f32
  unsigned u; __builtin_memcpy(&u, &r, 4); return u;
}
#else
__device__ __forceinline__ unsigned pk2(float a, float b){
  return (unsigned)f2bf(a) | ((unsigned)f2bf(b) << 16);
}
#endif

union Frag { unsigned u[8]; uint4 q[2]; bf16x16 v; };

__global__ __launch_bounds__(256)
void ra_fwd(const float* __restrict__ Q, const float* __restrict__ K,
            const float* __restrict__ V, const float* __restrict__ PK,
            const float* __restrict__ PV, const int* __restrict__ R2T,
            float* __restrict__ OUT)
{
  __shared__ unsigned short sK [KT * KROW];     // K tile  [key][d]   bf16
  __shared__ unsigned short sVT[ND * VROW];     // V tile  [d][key]   bf16 (transposed)
  __shared__ unsigned short sP [8 * 16 * PROW]; // per-wave P scratch [row][key]

  const int bid = blockIdx.x;
  const int qt  = bid & 15;            // E/QBLK = 16
  const int hkv = (bid >> 4) & 7;
  const int b   = bid >> 7;
  const int qb  = qt * QBLK;

  const int tid  = threadIdx.x;
  const int lane = tid & 31;
  const int wav  = tid >> 5;
  const int g    = wav >> 1;
  const int hq   = hkv * NG + g;
  const int q0   = qb + (wav & 1) * 16;
  const int ln   = lane & 15;
  const int hi   = lane >> 4;          // which 16-lane half
  const int off0 = hi ? 8  : 0;        // A-fragment K-offset per half
  const int koff = hi ? 8  : 0;        // B-fragment K-offset per half

  // ---- Q A-fragments (16 rows x 128 d), bf16, softmax scale pre-folded ----
  Frag qf[4];
  {
    const float* qrow = Q + ((size_t)((b*NE + q0 + ln)*NHQ + hq)) * ND;
    #pragma unroll
    for (int dc = 0; dc < 4; ++dc){
      const float* p0 = qrow + dc*32 + off0;
      float4 f0 = *(const float4*)(p0);
      float4 f1 = *(const float4*)(p0 + 4);
      float4 f2 = *(const float4*)(p0 + 16);
      float4 f3 = *(const float4*)(p0 + 20);
      qf[dc].u[0]=pk2(f0.x*QSCALE,f0.y*QSCALE); qf[dc].u[1]=pk2(f0.z*QSCALE,f0.w*QSCALE);
      qf[dc].u[2]=pk2(f1.x*QSCALE,f1.y*QSCALE); qf[dc].u[3]=pk2(f1.z*QSCALE,f1.w*QSCALE);
      qf[dc].u[4]=pk2(f2.x*QSCALE,f2.y*QSCALE); qf[dc].u[5]=pk2(f2.z*QSCALE,f2.w*QSCALE);
      qf[dc].u[6]=pk2(f3.x*QSCALE,f3.y*QSCALE); qf[dc].u[7]=pk2(f3.z*QSCALE,f3.w*QSCALE);
    }
  }

  // ---- online softmax state (replicated across each 16-lane half) ----
  float mrow[8], ssum[8];
  f32x8 acc[8];
  #pragma unroll
  for (int j = 0; j < 8; ++j){ mrow[j] = -1e30f; ssum[j] = 0.f; }
  #pragma unroll
  for (int dt = 0; dt < 8; ++dt)
    acc[dt] = (f32x8){0.f,0.f,0.f,0.f,0.f,0.f,0.f,0.f};

  const int pbase = wav * 16 * PROW;

  // Flattened tile list: causal extend tiles then paged-prefix tiles.
  // One online softmax over both == exact logaddexp merge of the two states.
  const int nkeysE = qb + QBLK;
  const int ntE    = (nkeysE + KT - 1) / KT;
  const int ntot   = ntE + NP / KT;

  const int key = tid >> 2;            // 0..63 : key row this thread stages
  const int seg = tid & 3;             // d range [seg*32, seg*32+32)
  float4 fK[8], fV[8];

  // issue global loads for tile i into registers (no wait here -> overlaps compute)
  auto issue_loads = [&](int i){
    const float *ks, *vs;
    if (i < ntE){
      int kpos = i*KT + key;
      if (kpos >= nkeysE) kpos = nkeysE - 1;     // tail rows are causally masked anyway
      const size_t base = ((size_t)(b*NE + kpos)*NHKV + hkv) * ND;
      ks = K + base; vs = V + base;
    } else {
      const int kpos = (i - ntE)*KT + key;
      const int tok  = R2T[b*NP + kpos];         // page-table gather
      const size_t base = ((size_t)tok*NHKV + hkv) * ND;
      ks = PK + base; vs = PV + base;
    }
    #pragma unroll
    for (int x = 0; x < 8; ++x) fK[x] = *(const float4*)(ks + seg*32 + x*4);
    #pragma unroll
    for (int x = 0; x < 8; ++x) fV[x] = *(const float4*)(vs + seg*32 + x*4);
  };

  // convert staged registers to bf16 and store into LDS (V transposed)
  auto store_tile = [&](){
    #pragma unroll
    for (int x = 0; x < 4; ++x){
      uint4 w;
      w.x = pk2(fK[2*x].x,   fK[2*x].y);   w.y = pk2(fK[2*x].z,   fK[2*x].w);
      w.z = pk2(fK[2*x+1].x, fK[2*x+1].y); w.w = pk2(fK[2*x+1].z, fK[2*x+1].w);
      *(uint4*)&sK[key*KROW + seg*32 + x*8] = w;
    }
    #pragma unroll
    for (int x = 0; x < 8; ++x){
      const int d = seg*32 + x*4;
      sVT[(d+0)*VROW + key] = f2bf(fV[x].x);
      sVT[(d+1)*VROW + key] = f2bf(fV[x].y);
      sVT[(d+2)*VROW + key] = f2bf(fV[x].z);
      sVT[(d+3)*VROW + key] = f2bf(fV[x].w);
    }
  };

  issue_loads(0);

  for (int i = 0; i < ntot; ++i){
    __syncthreads();                   // all waves done reading previous tile
    store_tile();                      // waits on this tile's global loads here
    if (i + 1 < ntot) issue_loads(i + 1);   // prefetch next tile (overlaps WMMAs)
    __syncthreads();                   // tile visible to all waves

    const bool causal = (i < ntE);
    const int  t0     = causal ? i*KT : (i - ntE)*KT;

    #pragma unroll
    for (int h = 0; h < 2; ++h){       // two 32-key blocks per tile
      float sc[2][8];
      #pragma unroll
      for (int nt = 0; nt < 2; ++nt){
        const int krow = h*32 + nt*16 + ln;   // key this lane serves (B-layout N=lane%16)
        // batch all 8 ds_load_b128 first -> partial dscnt waits under the WMMA chain
        Frag kb[4];
        #pragma unroll
        for (int dc = 0; dc < 4; ++dc){
          kb[dc].q[0] = *(uint4*)&sK[krow*KROW + dc*32 + koff];
          kb[dc].q[1] = *(uint4*)&sK[krow*KROW + dc*32 + koff + 16];
        }
        f32x8 s = (f32x8){0.f,0.f,0.f,0.f,0.f,0.f,0.f,0.f};
        #pragma unroll
        for (int dc = 0; dc < 4; ++dc)
          s = __builtin_amdgcn_wmma_f32_16x16x32_bf16(false, qf[dc].v, false, kb[dc].v,
                                                      (short)0, s, false, false);
        const int kpos = t0 + h*32 + nt*16 + ln;
        #pragma unroll
        for (int j = 0; j < 8; ++j){
          float val = s[j];            // scale already folded into Q
          if (causal){
            const int qpos = q0 + hi*8 + j;   // C-layout: row = j (+8 in upper half)
            if (kpos > qpos) val = -1e30f;    // causal mask (also kills tail rows)
          }
          sc[nt][j] = val;
        }
      }
      // row-wise online softmax over this 32-key block
      float fac[8], psum[8];
      #pragma unroll
      for (int j = 0; j < 8; ++j){
        float mt = fmaxf(sc[0][j], sc[1][j]);
        mt = fmaxf(mt, SWZ(mt,1)); mt = fmaxf(mt, SWZ(mt,2));
        mt = fmaxf(mt, SWZ(mt,4)); mt = fmaxf(mt, SWZ(mt,8));
        const float mn = fmaxf(mrow[j], mt);
        fac[j]  = __expf(mrow[j] - mn);
        mrow[j] = mn;
        const float p0 = __expf(sc[0][j] - mn);
        const float p1 = __expf(sc[1][j] - mn);
        const int row = hi*8 + j;
        sP[pbase + row*PROW +      ln] = f2bf(p0);   // C-layout -> A-layout via LDS
        sP[pbase + row*PROW + 16 + ln] = f2bf(p1);
        float ps = p0 + p1;
        ps += SWZ(ps,1); ps += SWZ(ps,2); ps += SWZ(ps,4); ps += SWZ(ps,8);
        psum[j] = ps;
      }

      // issue P A-fragment loads early: store->load dscnt wait hides under the
      // ssum/acc rescale VALU work below (DS ops are in-order within a wave)
      Frag pf;
      pf.q[0] = *(uint4*)&sP[pbase + ln*PROW + off0];
      pf.q[1] = *(uint4*)&sP[pbase + ln*PROW + 16 + off0];

      #pragma unroll
      for (int j = 0; j < 8; ++j) ssum[j] = ssum[j]*fac[j] + psum[j];
      #pragma unroll
      for (int dt = 0; dt < 8; ++dt)
        #pragma unroll
        for (int j = 0; j < 8; ++j) acc[dt][j] *= fac[j];

      // P x V with rolling one-stage V-fragment prefetch (acc[dt] are independent)
      Frag vb;
      vb.q[0] = *(uint4*)&sVT[(0*16 + ln)*VROW + h*32 + koff];
      vb.q[1] = *(uint4*)&sVT[(0*16 + ln)*VROW + h*32 + koff + 16];
      #pragma unroll
      for (int dt = 0; dt < 8; ++dt){
        Frag nxt;
        if (dt + 1 < 8){
          nxt.q[0] = *(uint4*)&sVT[((dt+1)*16 + ln)*VROW + h*32 + koff];
          nxt.q[1] = *(uint4*)&sVT[((dt+1)*16 + ln)*VROW + h*32 + koff + 16];
        }
        acc[dt] = __builtin_amdgcn_wmma_f32_16x16x32_bf16(false, pf.v, false, vb.v,
                                                          (short)0, acc[dt], false, false);
        if (dt + 1 < 8) vb = nxt;
      }
    }
  }

  // ---- normalize + write out[B*E, HQ*D] ----
  float inv[8];
  #pragma unroll
  for (int j = 0; j < 8; ++j) inv[j] = 1.0f / ssum[j];
  #pragma unroll
  for (int dt = 0; dt < 8; ++dt){
    #pragma unroll
    for (int j = 0; j < 8; ++j){
      const int qpos = q0 + hi*8 + j;
      OUT[((size_t)(b*NE + qpos))*(NHQ*ND) + hq*ND + dt*16 + ln] = acc[dt][j] * inv[j];
    }
  }
}

extern "C" void kernel_launch(void* const* d_in, const int* in_sizes, int n_in,
                              void* d_out, int out_size, void* d_ws, size_t ws_size,
                              hipStream_t stream) {
  const float* q  = (const float*)d_in[0];
  const float* k  = (const float*)d_in[1];
  const float* v  = (const float*)d_in[2];
  const float* pk = (const float*)d_in[3];
  const float* pv = (const float*)d_in[4];
  const int*  r2t = (const int*)d_in[5];
  float* out = (float*)d_out;
  (void)in_sizes; (void)n_in; (void)out_size; (void)d_ws; (void)ws_size;
  dim3 grid(NB * NHKV * (NE / QBLK));   // 4 * 8 * 16 = 512 workgroups, 8 waves each
  ra_fwd<<<grid, 256, 0, stream>>>(q, k, v, pk, pv, r2t, out);
}
